// AutoencoderDecoder_61254823576231
// MI455X (gfx1250) — matrix operand order repaired and checked by reading, not compiled, not run on hardware
//
#include <hip/hip_runtime.h>
#include <hip/hip_bf16.h>
#include <math.h>

typedef _Float16 half_t;
typedef __attribute__((ext_vector_type(16))) _Float16 v16h;
typedef __attribute__((ext_vector_type(4)))  _Float16 v4h;
typedef __attribute__((ext_vector_type(8)))  float    v8f;
typedef __attribute__((ext_vector_type(4)))  unsigned int v4u;
typedef __attribute__((ext_vector_type(8)))  int      v8i;
typedef __attribute__((ext_vector_type(4)))  int      v4i;

#define BB   2
#define SS   1800
#define DD   128
#define HH   8
#define DK   16
#define DFF  512
#define LL   6
#define VV   16
#define LAT  1024
#define MTOT (BB * SS)          // 3600 rows
#define COLS (SS * DD)          // 230400

// ---------------------------------------------------------------------------
// Latent projection: h = relu(latent @ lp_w1 + lp_b1)
// ---------------------------------------------------------------------------
__global__ void latent_kernel(const float* __restrict__ latent,
                              const float* __restrict__ w1,
                              const float* __restrict__ b1,
                              float* __restrict__ hout) {
    int j = threadIdx.x;
    float a0 = 0.f, a1 = 0.f;
    for (int k = 0; k < LAT; ++k) {
        float w = w1[(size_t)k * 256 + j];
        a0 += latent[k] * w;
        a1 += latent[LAT + k] * w;
    }
    hout[j]       = fmaxf(a0 + b1[j], 0.f);
    hout[256 + j] = fmaxf(a1 + b1[j], 0.f);
}

// ---------------------------------------------------------------------------
// Weight convert + transpose: WT[z][n][k] = (f16) W[z][k][n]
// ---------------------------------------------------------------------------
__global__ void convtrans_kernel(const float* __restrict__ W,
                                 half_t* __restrict__ WT, int K, int N) {
    size_t mat = (size_t)blockIdx.z * K * N;
    int idx = blockIdx.x * 256 + threadIdx.x;
    if (idx >= K * N) return;
    int k = idx / N, n = idx - k * N;
    WT[mat + (size_t)n * K + k] = (half_t)W[mat + idx];
}

// ---------------------------------------------------------------------------
// dec = h @ lp_w2 + lp_b2 + pos_encoding ; float4 per thread (4 columns)
// grid 225 x 256
// ---------------------------------------------------------------------------
__global__ void decinit_kernel(const float* __restrict__ h,
                               const float* __restrict__ w2,
                               const float* __restrict__ b2,
                               float* __restrict__ X,
                               half_t* __restrict__ Xh,
                               half_t* __restrict__ Eh) {
    __shared__ float hs[512];
    int t = threadIdx.x;
    hs[t]       = h[t];
    hs[256 + t] = h[256 + t];
    __syncthreads();
    size_t c0 = ((size_t)blockIdx.x * 256 + t) * 4;
    float a0[4] = {0, 0, 0, 0}, a1[4] = {0, 0, 0, 0};
    for (int k = 0; k < 256; ++k) {
        const float4 w = *(const float4*)(w2 + (size_t)k * COLS + c0);
        float h0 = hs[k], h1 = hs[256 + k];
        a0[0] += h0 * w.x; a0[1] += h0 * w.y; a0[2] += h0 * w.z; a0[3] += h0 * w.w;
        a1[0] += h1 * w.x; a1[1] += h1 * w.y; a1[2] += h1 * w.z; a1[3] += h1 * w.w;
    }
    int s = (int)(c0 / DD);
    float4 o0, o1; v4h q0, q1;
    float* o0p = &o0.x; float* o1p = &o1.x;
#pragma unroll
    for (int j = 0; j < 4; ++j) {
        int d = (int)((c0 + j) & (DD - 1));
        float freq = expf((float)(d & ~1) * (-9.2103403719761836f / (float)DD));
        float ang  = (float)s * freq;
        float pe   = (d & 1) ? cosf(ang) : sinf(ang);
        float bias = b2[c0 + j];
        float v0 = a0[j] + bias + pe;
        float v1 = a1[j] + bias + pe;
        o0p[j] = v0; o1p[j] = v1;
        q0[j] = (half_t)v0; q1[j] = (half_t)v1;
    }
    *(float4*)(X + c0)         = o0;
    *(float4*)(X + COLS + c0)  = o1;
    *(v4h*)(Xh + c0)           = q0;
    *(v4h*)(Xh + COLS + c0)    = q1;
    *(v4h*)(Eh + c0)           = q0;
    *(v4h*)(Eh + COLS + c0)    = q1;
}

// ---------------------------------------------------------------------------
// WMMA GEMM with register double-buffering over K chunks.
// ---------------------------------------------------------------------------
__global__ __launch_bounds__(32)
void gemm_wmma_kernel(const half_t* __restrict__ A,
                      const half_t* __restrict__ WT,
                      const float*  __restrict__ bias,
                      float*  __restrict__ Cf,
                      half_t* __restrict__ Ch,
                      int M, int N, int K, int relu) {
    int lane = threadIdx.x & 31;
    int hi   = lane >> 4;
    int ln   = lane & 15;
    int mt   = blockIdx.x;
    int nt   = blockIdx.y;
    int m    = mt * 16 + ln;
    int nn   = nt * 16 + ln;

    const half_t* Ap = A  + (size_t)m  * K + hi * 16;
    const half_t* Wp = WT + (size_t)nn * K + hi * 16;

    auto loadfrag = [&](v16h& a, v16h& bb, int k0) {
#pragma unroll
        for (int i = 0; i < 8; ++i) {
            a[i]     = Ap[k0 + i];
            a[8 + i] = Ap[k0 + 8 + i];
        }
#pragma unroll
        for (int e = 0; e < 16; ++e)
            bb[e] = Wp[k0 + e];
    };

    v8f acc = {};
    v16h a0, b0, a1, b1;
    loadfrag(a0, b0, 0);
    int k0 = 0;
    for (; k0 + 32 < K; k0 += 32) {
        loadfrag(a1, b1, k0 + 32);                 // overlap with WMMA below
        __builtin_prefetch(Ap + k0 + 64, 0, 1);
        acc = __builtin_amdgcn_wmma_f32_16x16x32_f16(
            false, a0, false, b0, (short)0, acc, false, false);
        a0 = a1; b0 = b1;
    }
    acc = __builtin_amdgcn_wmma_f32_16x16x32_f16(
        false, a0, false, b0, (short)0, acc, false, false);

    float bv = bias ? bias[nn] : 0.f;
    float out[8];
#pragma unroll
    for (int r = 0; r < 8; ++r) {
        float v = acc[r] + bv;
        out[r] = relu ? fmaxf(v, 0.f) : v;
    }
    int mbase = mt * 16 + 8 * hi;
    if (Cf) {
#pragma unroll
        for (int r = 0; r < 8; ++r)
            Cf[(size_t)(mbase + r) * N + nn] = out[r];
    }
    if (Ch) {
#pragma unroll
        for (int r = 0; r < 8; ++r)
            Ch[(size_t)(mbase + r) * N + nn] = (half_t)out[r];
    }
}

// ---------------------------------------------------------------------------
// Async copy of 16 bytes (lane-addressed) global -> LDS
// ---------------------------------------------------------------------------
__device__ __forceinline__ void async_ld128(unsigned ldsoff, const void* gptr) {
    unsigned long long ga = (unsigned long long)(size_t)gptr;
    asm volatile("global_load_async_to_lds_b128 %0, %1, off"
                 :: "v"(ldsoff), "v"(ga) : "memory");
}

// ---------------------------------------------------------------------------
// TDM: load a [rows<=32] x 16-half 2D tile (row stride 128 halves) into LDS.
// D# packed per CDNA5 ISA 8.3/8.4. All operands wave-uniform.
// ---------------------------------------------------------------------------
__device__ __forceinline__ void tdm_load_tile(unsigned ldsoff, const half_t* gp,
                                              int rows) {
    unsigned long long ga = (unsigned long long)(size_t)gp;
    v4u g0;
    g0[0] = 1u;                                   // count=1, user descriptor
    g0[1] = ldsoff;                               // lds_addr
    g0[2] = (unsigned)ga;                         // global_addr[31:0]
    g0[3] = (unsigned)(ga >> 32) | (2u << 30);    // global_addr[56:32] | type=2
    v8i g1;
    g1[0] = 0x00010000;                           // data_size=2 bytes
    g1[1] = (int)(16u << 16);                     // tensor_dim0 = 16
    g1[2] = (int)(((unsigned)rows & 0xFFFFu) << 16);  // tensor_dim1 lo
    g1[3] = (int)((16u << 16) | (((unsigned)rows >> 16) & 0xFFFFu)); // tile_dim0=16
    g1[4] = 32;                                   // tile_dim1 = 32
    g1[5] = DD;                                   // tensor_dim0_stride = 128
    g1[6] = 0;
    g1[7] = 0;
    v4i g2 = {}; v4i g3 = {};
#if __clang_major__ >= 23
    v8i g4 = {};
    __builtin_amdgcn_tensor_load_to_lds(g0, g1, g2, g3, g4, 0);
#else
    __builtin_amdgcn_tensor_load_to_lds(g0, g1, g2, g3, 0);
#endif
}

// ---------------------------------------------------------------------------
// Fused flash-style attention: TDM-staged K, async-staged V, double buffered.
// ---------------------------------------------------------------------------
__global__ __launch_bounds__(32)
void attn_kernel(const half_t* __restrict__ Qb,
                 const half_t* __restrict__ Kb,
                 const half_t* __restrict__ Vb,
                 half_t* __restrict__ Ob,
                 int causal) {
    __shared__ __align__(16) half_t kbuf[2][32 * DK];
    __shared__ __align__(16) half_t vbuf[2][32 * DK];
    __shared__ __align__(16) half_t pl[16 * 32];
    __shared__ __align__(16) half_t zpad[16];     // 32B of zeros
    int lane = threadIdx.x & 31;
    int hi   = lane >> 4;
    int ln   = lane & 15;
    int qt = blockIdx.x, hd = blockIdx.y, b = blockIdx.z;
    int koff = hi * 8;
    const float NEG_INF = -__builtin_inff();

    if (lane < 8) ((unsigned*)zpad)[lane] = 0u;
    asm volatile("s_wait_dscnt 0" ::: "memory");

    // Q fragment (A layout, K padded: dk 0..15 valid, 16..31 zero)
    int qrow = qt * 16 + ln;
    int qrc  = qrow < SS ? qrow : (SS - 1);
    const half_t* Qp = Qb + ((size_t)(b * SS + qrc)) * DD + hd * 16;
    v16h aq;
#pragma unroll
    for (int i = 0; i < 8; ++i) { aq[i] = hi ? (half_t)0 : Qp[i]; aq[8 + i] = hi ? (half_t)0 : Qp[8 + i]; }

    auto stage = [&](int j0, int buf) {
        // K chunk via Tensor Data Mover (OOB rows -> zero)
        tdm_load_tile((unsigned)(size_t)&kbuf[buf][0],
                      Kb + ((size_t)(b * SS) + j0) * DD + hd * 16, SS - j0);
        // V chunk via async lane-addressed copies (clamped rows)
        int key = j0 + lane;
        key = key < SS ? key : (SS - 1);
        const half_t* vg = Vb + ((size_t)(b * SS + key)) * DD + hd * 16;
        unsigned vo = (unsigned)(size_t)&vbuf[buf][lane * DK];
        async_ld128(vo,      vg);
        async_ld128(vo + 16, vg + 8);
    };

    float rmax[8], rsum[8];
    v8f acc = {};
#pragma unroll
    for (int r = 0; r < 8; ++r) { rmax[r] = NEG_INF; rsum[r] = 0.f; }

    stage(0, 0);
    for (int j0 = 0; j0 < SS; j0 += 32) {
        int cur = (j0 >> 5) & 1;
        int nxt = j0 + 32;
        if (nxt < SS) {
            stage(nxt, cur ^ 1);
            __builtin_amdgcn_s_wait_tensorcnt(1);
            asm volatile("s_wait_asynccnt 2" ::: "memory");
        } else {
            __builtin_amdgcn_s_wait_tensorcnt(0);
            asm volatile("s_wait_asynccnt 0" ::: "memory");
        }
        const half_t* kd = kbuf[cur];
        const half_t* vd = vbuf[cur];

        // K fragments: pointer-select zero pad for dk rows 16..31
        const half_t* k0p = hi ? zpad : kd + ln * DK;
        const half_t* k1p = hi ? zpad : kd + (16 + ln) * DK;
        v16h bk0 = *(const v16h*)k0p;
        v16h bk1 = *(const v16h*)k1p;
        v8f s0 = {}, s1 = {};
        s0 = __builtin_amdgcn_wmma_f32_16x16x32_f16(false, aq, false, bk0,
                                                    (short)0, s0, false, false);
        s1 = __builtin_amdgcn_wmma_f32_16x16x32_f16(false, aq, false, bk1,
                                                    (short)0, s1, false, false);
#pragma unroll
        for (int r = 0; r < 8; ++r) {
            int q   = qt * 16 + r + 8 * hi;
            int k0i = j0 + ln;
            int k1i = j0 + 16 + ln;
            float v0 = s0[r] * 0.25f;   // 1/sqrt(DK)
            float v1 = s1[r] * 0.25f;
            if (causal && !(k0i > q)) v0 = -1e9f;
            if (causal && !(k1i > q)) v1 = -1e9f;
            if (k0i >= SS) v0 = NEG_INF;
            if (k1i >= SS) v1 = NEG_INF;
            float m2 = fmaxf(v0, v1);
#pragma unroll
            for (int off = 8; off; off >>= 1)
                m2 = fmaxf(m2, __shfl_xor(m2, off, 32));
            float nm    = fmaxf(rmax[r], m2);
            float alpha = __expf(rmax[r] - nm);
            float p0 = __expf(v0 - nm);
            float p1 = __expf(v1 - nm);
            float ps = p0 + p1;
#pragma unroll
            for (int off = 8; off; off >>= 1)
                ps += __shfl_xor(ps, off, 32);
            rsum[r] = rsum[r] * alpha + ps;
            rmax[r] = nm;
            acc[r] *= alpha;
            pl[(r + 8 * hi) * 32 + ln]      = (half_t)p0;
            pl[(r + 8 * hi) * 32 + 16 + ln] = (half_t)p1;
        }
        asm volatile("s_wait_dscnt 0" ::: "memory");
        // P fragment (A layout, 16x32) from LDS
        const half_t* pp = pl + ln * 32 + koff;
        v16h ap;
#pragma unroll
        for (int i = 0; i < 8; ++i) {
            ap[i]     = pp[i];
            ap[8 + i] = pp[16 + i];
        }
        // V fragment (B layout, 32x16) from LDS
        v16h bv;
        int kb = hi * 8;
#pragma unroll
        for (int e = 0; e < 16; ++e)
            bv[e] = vd[(kb + e + (e >= 8 ? 8 : 0)) * DK + ln];
        acc = __builtin_amdgcn_wmma_f32_16x16x32_f16(false, ap, false, bv,
                                                     (short)0, acc, false, false);
    }
#pragma unroll
    for (int r = 0; r < 8; ++r) {
        int q = qt * 16 + r + 8 * hi;
        if (q < SS)
            Ob[((size_t)(b * SS + q)) * DD + hd * 16 + ln] =
                (half_t)(acc[r] / rsum[r]);
    }
}

// ---------------------------------------------------------------------------
// LayerNorm with fused residual, in-place capable.
// ---------------------------------------------------------------------------
__global__ __launch_bounds__(32)
void ln_kernel(const float* __restrict__ X, const float* __restrict__ R,
               const float* __restrict__ g, const float* __restrict__ bta,
               float* __restrict__ Y, half_t* __restrict__ Yh) {
    int row  = blockIdx.x;
    int lane = threadIdx.x;
    float v[4];
#pragma unroll
    for (int i = 0; i < 4; ++i) {
        int d = lane + i * 32;
        size_t idx = (size_t)row * DD + d;
        v[i] = X[idx] + R[idx];
    }
    float s = v[0] + v[1] + v[2] + v[3];
#pragma unroll
    for (int off = 16; off; off >>= 1) s += __shfl_xor(s, off, 32);
    float mu = s / (float)DD;
    float q = 0.f;
#pragma unroll
    for (int i = 0; i < 4; ++i) { float t = v[i] - mu; q += t * t; }
#pragma unroll
    for (int off = 16; off; off >>= 1) q += __shfl_xor(q, off, 32);
    float rstd = rsqrtf(q / (float)DD + 1e-5f);
#pragma unroll
    for (int i = 0; i < 4; ++i) {
        int d = lane + i * 32;
        size_t idx = (size_t)row * DD + d;
        float o = (v[i] - mu) * rstd * g[d] + bta[d];
        Y[idx]  = o;
        Yh[idx] = (half_t)o;
    }
}

// ---------------------------------------------------------------------------
extern "C" void kernel_launch(void* const* d_in, const int* in_sizes, int n_in,
                              void* d_out, int out_size, void* d_ws, size_t ws_size,
                              hipStream_t stream) {
    const float* latent = (const float*)d_in[0];
    const float* lp_w1  = (const float*)d_in[1];
    const float* lp_b1  = (const float*)d_in[2];
    const float* lp_w2  = (const float*)d_in[3];
    const float* lp_b2  = (const float*)d_in[4];
    const float* out_w  = (const float*)d_in[5];
    const float* out_b  = (const float*)d_in[6];
    const float* sa_wq = (const float*)d_in[7],  *sa_bq = (const float*)d_in[8];
    const float* sa_wk = (const float*)d_in[9],  *sa_bk = (const float*)d_in[10];
    const float* sa_wv = (const float*)d_in[11], *sa_bv = (const float*)d_in[12];
    const float* sa_wo = (const float*)d_in[13], *sa_bo = (const float*)d_in[14];
    const float* ca_wq = (const float*)d_in[15], *ca_bq = (const float*)d_in[16];
    const float* ca_wk = (const float*)d_in[17], *ca_bk = (const float*)d_in[18];
    const float* ca_wv = (const float*)d_in[19], *ca_bv = (const float*)d_in[20];
    const float* ca_wo = (const float*)d_in[21], *ca_bo = (const float*)d_in[22];
    const float* ff_w1 = (const float*)d_in[23], *ff_b1 = (const float*)d_in[24];
    const float* ff_w2 = (const float*)d_in[25], *ff_b2 = (const float*)d_in[26];
    const float* ln1_g = (const float*)d_in[27], *ln1_b = (const float*)d_in[28];
    const float* ln2_g = (const float*)d_in[29], *ln2_b = (const float*)d_in[30];
    const float* ln3_g = (const float*)d_in[31], *ln3_b = (const float*)d_in[32];

    size_t off = 0;
    auto carve = [&](size_t bytes) -> void* {
        void* p = (char*)d_ws + off;
        off += (bytes + 255) & ~(size_t)255;
        return p;
    };
    float*  hbuf = (float*)carve(512 * sizeof(float));
    float*  X    = (float*)carve((size_t)BB * COLS * sizeof(float));
    half_t* Xh   = (half_t*)carve((size_t)BB * COLS * sizeof(half_t));
    half_t* Eh   = (half_t*)carve((size_t)BB * COLS * sizeof(half_t));
    half_t* Qh   = (half_t*)carve((size_t)BB * COLS * sizeof(half_t));
    half_t* Kh   = (half_t*)carve((size_t)BB * COLS * sizeof(half_t));
    half_t* Vh   = (half_t*)carve((size_t)BB * COLS * sizeof(half_t));
    half_t* AOh  = (half_t*)carve((size_t)BB * COLS * sizeof(half_t));
    float*  Proj = (float*)carve((size_t)BB * COLS * sizeof(float));
    half_t* F1h  = (half_t*)carve((size_t)MTOT * DFF * sizeof(half_t));
    size_t dd = (size_t)LL * DD * DD, df = (size_t)LL * DD * DFF;
    half_t* sa_wqT = (half_t*)carve(dd * 2); half_t* sa_wkT = (half_t*)carve(dd * 2);
    half_t* sa_wvT = (half_t*)carve(dd * 2); half_t* sa_woT = (half_t*)carve(dd * 2);
    half_t* ca_wqT = (half_t*)carve(dd * 2); half_t* ca_wkT = (half_t*)carve(dd * 2);
    half_t* ca_wvT = (half_t*)carve(dd * 2); half_t* ca_woT = (half_t*)carve(dd * 2);
    half_t* ff_w1T = (half_t*)carve(df * 2); half_t* ff_w2T = (half_t*)carve(df * 2);
    half_t* out_wT = (half_t*)carve((size_t)DD * VV * 2);

    dim3 gcDD((DD * DD + 255) / 256, 1, LL);
    dim3 gcDF((DD * DFF + 255) / 256, 1, LL);
    convtrans_kernel<<<gcDD, 256, 0, stream>>>(sa_wq, sa_wqT, DD, DD);
    convtrans_kernel<<<gcDD, 256, 0, stream>>>(sa_wk, sa_wkT, DD, DD);
    convtrans_kernel<<<gcDD, 256, 0, stream>>>(sa_wv, sa_wvT, DD, DD);
    convtrans_kernel<<<gcDD, 256, 0, stream>>>(sa_wo, sa_woT, DD, DD);
    convtrans_kernel<<<gcDD, 256, 0, stream>>>(ca_wq, ca_wqT, DD, DD);
    convtrans_kernel<<<gcDD, 256, 0, stream>>>(ca_wk, ca_wkT, DD, DD);
    convtrans_kernel<<<gcDD, 256, 0, stream>>>(ca_wv, ca_wvT, DD, DD);
    convtrans_kernel<<<gcDD, 256, 0, stream>>>(ca_wo, ca_woT, DD, DD);
    convtrans_kernel<<<gcDF, 256, 0, stream>>>(ff_w1, ff_w1T, DD, DFF);
    convtrans_kernel<<<gcDF, 256, 0, stream>>>(ff_w2, ff_w2T, DFF, DD);
    convtrans_kernel<<<dim3(8, 1, 1), 256, 0, stream>>>(out_w, out_wT, DD, VV);

    dim3 gD(MTOT / 16, DD / 16);
    dim3 gF1(MTOT / 16, DFF / 16);
    dim3 gOut(MTOT / 16, VV / 16);
    dim3 gAttn((SS + 15) / 16, HH, BB);

    latent_kernel<<<1, 256, 0, stream>>>(latent, lp_w1, lp_b1, hbuf);
    decinit_kernel<<<COLS / 1024, 256, 0, stream>>>(hbuf, lp_w2, lp_b2, X, Xh, Eh);

    for (int i = 0; i < LL; ++i) {
        size_t wOff = (size_t)i * DD * DD, bOff = (size_t)i * DD;
        gemm_wmma_kernel<<<gD, 32, 0, stream>>>(Xh, sa_wqT + wOff, sa_bq + bOff,
                                                nullptr, Qh, MTOT, DD, DD, 0);
        gemm_wmma_kernel<<<gD, 32, 0, stream>>>(Xh, sa_wkT + wOff, sa_bk + bOff,
                                                nullptr, Kh, MTOT, DD, DD, 0);
        gemm_wmma_kernel<<<gD, 32, 0, stream>>>(Xh, sa_wvT + wOff, sa_bv + bOff,
                                                nullptr, Vh, MTOT, DD, DD, 0);
        attn_kernel<<<gAttn, 32, 0, stream>>>(Qh, Kh, Vh, AOh, 1);
        gemm_wmma_kernel<<<gD, 32, 0, stream>>>(AOh, sa_woT + wOff, sa_bo + bOff,
                                                Proj, nullptr, MTOT, DD, DD, 0);
        ln_kernel<<<MTOT, 32, 0, stream>>>(X, Proj, ln1_g + bOff, ln1_b + bOff, X, Xh);
        gemm_wmma_kernel<<<gD, 32, 0, stream>>>(Xh, ca_wqT + wOff, ca_bq + bOff,
                                                nullptr, Qh, MTOT, DD, DD, 0);
        gemm_wmma_kernel<<<gD, 32, 0, stream>>>(Eh, ca_wkT + wOff, ca_bk + bOff,
                                                nullptr, Kh, MTOT, DD, DD, 0);
        gemm_wmma_kernel<<<gD, 32, 0, stream>>>(Eh, ca_wvT + wOff, ca_bv + bOff,
                                                nullptr, Vh, MTOT, DD, DD, 0);
        attn_kernel<<<gAttn, 32, 0, stream>>>(Qh, Kh, Vh, AOh, 0);
        gemm_wmma_kernel<<<gD, 32, 0, stream>>>(AOh, ca_woT + wOff, ca_bo + bOff,
                                                Proj, nullptr, MTOT, DD, DD, 0);
        ln_kernel<<<MTOT, 32, 0, stream>>>(X, Proj, ln2_g + bOff, ln2_b + bOff, X, Xh);
        gemm_wmma_kernel<<<gF1, 32, 0, stream>>>(Xh, ff_w1T + (size_t)i * DD * DFF,
                                                 ff_b1 + (size_t)i * DFF,
                                                 nullptr, F1h, MTOT, DFF, DD, 1);
        gemm_wmma_kernel<<<gD, 32, 0, stream>>>(F1h, ff_w2T + (size_t)i * DD * DFF,
                                                ff_b2 + bOff,
                                                Proj, nullptr, MTOT, DD, DFF, 0);
        ln_kernel<<<MTOT, 32, 0, stream>>>(X, Proj, ln3_g + bOff, ln3_b + bOff, X, Xh);
    }
    gemm_wmma_kernel<<<gOut, 32, 0, stream>>>(Xh, out_wT, out_b,
                                              (float*)d_out, nullptr, MTOT, VV, DD, 0);
}
